// MambaHierarchicalBlindScan_47888885350505
// MI455X (gfx1250) — matrix operand order repaired and checked by reading, not compiled, not run on hardware
//
#include <hip/hip_runtime.h>

// ---------------------------------------------------------------------------
// Types for CDNA5 WMMA (wave32): v_wmma_f32_16x16x32_bf16
// ---------------------------------------------------------------------------
typedef __bf16 bf16_t;
typedef __attribute__((ext_vector_type(16))) __bf16 v16bf;
typedef __attribute__((ext_vector_type(8)))  float  v8f;

#define HW_PIX 25600  // 160*160

// ---------------------------------------------------------------------------
// CDNA5 async global->LDS copy (GLOBAL_LOAD_ASYNC_TO_LDS_B128, ASYNCcnt).
// Toolchain signature (from clang diagnostics): (int4 AS1*, int4 AS3*,
// imm offset, imm cpol) -- non-const pointers. Guarded by __has_builtin.
// ---------------------------------------------------------------------------
#if defined(__has_builtin)
#  if __has_builtin(__builtin_amdgcn_global_load_async_to_lds_b128)
#    define HAVE_ASYNC_LDS 1
#  endif
#  if __has_builtin(__builtin_amdgcn_s_wait_asynccnt)
#    define HAVE_WAIT_ASYNC_BUILTIN 1
#  endif
#endif

typedef int int4v __attribute__((__vector_size__(16)));
typedef __attribute__((address_space(1))) int4v* as1_i4p;
typedef __attribute__((address_space(3))) int4v* as3_i4p;

static __device__ __forceinline__ void async_wait_all() {
#if defined(HAVE_WAIT_ASYNC_BUILTIN)
    __builtin_amdgcn_s_wait_asynccnt(0);
#else
    asm volatile("s_wait_asynccnt 0x0" ::: "memory");
#endif
}

static __device__ __forceinline__ float softplusf_(float x) {
    return (x > 15.0f) ? x : log1pf(__expf(x));
}
static __device__ __forceinline__ float siluf_(float x) {
    return x / (1.0f + __expf(-x));
}

// ---------------------------------------------------------------------------
// Fragment loaders per CDNA5 ISA VGPR layouts (CDNA5 ISA §7.12.2):
//   16-bit A 16x32: lane m = l&15; hi=l>>4; elem e<8 -> K=8*hi+e; e>=8 -> K=16+8*hi+(e-8)
//   16-bit B 32x16: lane n = l&15; hi=l>>4; elem e<8 -> K=8*hi+e; e>=8 -> K=16+8*hi+(e-8)
//   f32 C/D 16x16 : lane n = l&15; hi=l>>4; vgpr v   -> M=v+8*hi
// ---------------------------------------------------------------------------
static __device__ __forceinline__ v16bf frag_a(const bf16_t* base, int ld,
                                               int m0, int k0, int lane) {
    const int hi = lane >> 4;
    const bf16_t* p = base + (size_t)(m0 + (lane & 15)) * ld + k0;
    v16bf a;
#pragma unroll
    for (int e = 0; e < 8; ++e) a[e] = p[8 * hi + e];
#pragma unroll
    for (int e = 0; e < 8; ++e) a[8 + e] = p[16 + 8 * hi + e];
    return a;
}

// B tile of W^T where W is (Nreal x Kdim) f32 row-major (the *_w matrices):
// B[k][n] = W[n][k]; each lane reads two runs of 8 f32 (K=8*hi.., 16+8*hi..) and converts to bf16.
static __device__ __forceinline__ v16bf frag_b(const float* W, int Kdim,
                                               int n0, int k0, int Nreal, int lane) {
    const int n  = n0 + (lane & 15);
    const int hi = lane >> 4;
    v16bf b;
    if (n < Nreal) {
        const float* p = W + (size_t)n * Kdim + k0;
#pragma unroll
        for (int e = 0; e < 8; ++e) b[e] = (bf16_t)p[8 * hi + e];
#pragma unroll
        for (int e = 0; e < 8; ++e) b[8 + e] = (bf16_t)p[16 + 8 * hi + e];
    } else {
#pragma unroll
        for (int e = 0; e < 16; ++e) b[e] = (bf16_t)0.0f;
    }
    return b;
}

// ---------------------------------------------------------------------------
// Stage 1: 1x1 conv (32->96) + patchify + 8x8 snake + LayerNorm + shift
// -> xl (1600, 64, 96) bf16, ready as WMMA A-matrix rows.
// grid 1600 blocks (one per sequence) x 64 threads (one per seq position)
// ---------------------------------------------------------------------------
__global__ __launch_bounds__(64)
void k_stage1(const float* __restrict__ x, const float* __restrict__ Win,
              const float* __restrict__ bin, const float* __restrict__ g,
              const float* __restrict__ be, bf16_t* __restrict__ xl) {
    const int seq = blockIdx.x;          // b*400 + ph*20 + pw
    const int j   = threadIdx.x;         // sequence position after shift
    bf16_t* dst = xl + ((size_t)seq * 64 + j) * 96;
    if (j == 0) {
#pragma unroll
        for (int d = 0; d < 96; ++d) dst[d] = (bf16_t)0.0f;
        return;
    }
    // token j holds LN(xproj) of snake position j-1
    const int p  = j - 1;
    const int rr = p >> 3, cc = p & 7;
    const int q  = (rr & 1) ? (rr * 8 + (7 - cc)) : p;   // flat index in patch
    const int hh = q >> 3, ww = q & 7;
    const int b  = seq / 400;
    const int pr = seq % 400;
    const int ph = pr / 20, pw = pr % 20;
    const int h = ph * 8 + hh, w = pw * 8 + ww;

    const float* xp = x + (size_t)b * 32 * HW_PIX + h * 160 + w;
    float xin[32];
#pragma unroll
    for (int c = 0; c < 32; ++c) xin[c] = xp[(size_t)c * HW_PIX];

    float v[96];
    float mean = 0.0f;
#pragma unroll 4
    for (int d = 0; d < 96; ++d) {
        float s = bin[d];
        const float* wr = Win + d * 32;
#pragma unroll
        for (int c = 0; c < 32; ++c) s += wr[c] * xin[c];
        v[d] = s;
        mean += s;
    }
    mean *= (1.0f / 96.0f);
    float var = 0.0f;
#pragma unroll 4
    for (int d = 0; d < 96; ++d) { float t = v[d] - mean; var += t * t; }
    var *= (1.0f / 96.0f);
    const float inv = rsqrtf(var + 1e-5f);
#pragma unroll 4
    for (int d = 0; d < 96; ++d)
        dst[d] = (bf16_t)((v[d] - mean) * inv * g[d] + be[d]);
}

// ---------------------------------------------------------------------------
// Stage 2: fused local Mamba. One block (8 waves) per sequence.
// LDS (60 KB): xl[64][96] bf16 (later overlaid by dbl[64][48] bf16),
//              xcb[64][192] bf16 (xc -> xconv -> y, in place),
//              zb[64][192] bf16.
// ---------------------------------------------------------------------------
__global__ __launch_bounds__(256)
void k_local(const bf16_t* __restrict__ xl_g, const float* __restrict__ in_w,
             const float* __restrict__ conv_w, const float* __restrict__ conv_b,
             const float* __restrict__ xproj_w, const float* __restrict__ dt_w,
             const float* __restrict__ dt_b, const float* __restrict__ A_log,
             const float* __restrict__ Dv, const float* __restrict__ out_w,
             float* __restrict__ yl_out) {
    __shared__ __align__(16) unsigned char smem[61440];
    bf16_t* xl  = (bf16_t*)smem;                       // [64][96]
    bf16_t* dbl = (bf16_t*)smem;                       // [64][48] (after xl dead)
    bf16_t* xcb = (bf16_t*)(smem + 12288);             // [64][192]
    bf16_t* zb  = (bf16_t*)(smem + 12288 + 24576);     // [64][192]

    const int tid  = threadIdx.x;
    const int lane = tid & 31;
    const int wave = tid >> 5;
    const size_t seq = blockIdx.x;

    __builtin_prefetch(in_w, 0, 3);
    __builtin_prefetch(out_w, 0, 3);

    // ---- phase 0: stage xl tokens into LDS (12288 B) ----
#if defined(HAVE_ASYNC_LDS)
    {
        // GLOBAL_LOAD_ASYNC_TO_LDS_B128: each lane copies 16 B directly into
        // LDS, bypassing VGPRs; tracked by ASYNCcnt.
        char* gsrc = (char*)(xl_g + seq * (64 * 96));
#pragma unroll
        for (int i = 0; i < 3; ++i) {
            const int off = (tid + 256 * i) * 16;
            __builtin_amdgcn_global_load_async_to_lds_b128(
                (as1_i4p)(void*)(gsrc + off),
                (as3_i4p)(void*)(smem + off), 0, 0);
        }
        async_wait_all();
    }
#else
    {
        const uint4* src = (const uint4*)(xl_g + seq * (64 * 96));
        uint4* dst = (uint4*)xl;
#pragma unroll
        for (int i = 0; i < 3; ++i) dst[tid + 256 * i] = src[tid + 256 * i];
    }
#endif
    __syncthreads();

    // ---- phase 1: xz = xl @ in_w^T  (M=64, K=96, N=384) ----
    for (int tl = wave; tl < 96; tl += 8) {
        const int m0 = (tl & 3) << 4;
        const int n0 = (tl >> 2) << 4;
        v8f acc = {};
#pragma unroll
        for (int kk = 0; kk < 3; ++kk) {
            v16bf a = frag_a(xl, 96, m0, kk * 32, lane);
            v16bf b = frag_b(in_w, 96, n0, kk * 32, 384, lane);
            acc = __builtin_amdgcn_wmma_f32_16x16x32_bf16(false, a, false, b,
                                                          (short)0, acc, false, false);
        }
        const int n  = n0 + (lane & 15);
        const int hi = lane >> 4;
        if (n0 < 192) {
#pragma unroll
            for (int v = 0; v < 8; ++v)
                xcb[(m0 + v + 8 * hi) * 192 + n] = (bf16_t)acc[v];
        } else {
#pragma unroll
            for (int v = 0; v < 8; ++v)
                zb[(m0 + v + 8 * hi) * 192 + (n - 192)] = (bf16_t)acc[v];
        }
    }
    __syncthreads();

    // ---- phase 2: causal depthwise conv (dc=4) + SiLU, in place on xcb ----
    if (tid < 192) {
        const int d = tid;
        const float w0 = conv_w[d * 4 + 0], w1 = conv_w[d * 4 + 1];
        const float w2 = conv_w[d * 4 + 2], w3 = conv_w[d * 4 + 3];
        const float bi = conv_b[d];
        float xm1 = 0.0f, xm2 = 0.0f, xm3 = 0.0f;
        for (int t = 0; t < 64; ++t) {
            const float cur = (float)xcb[t * 192 + d];
            const float a = bi + w3 * cur + w2 * xm1 + w1 * xm2 + w0 * xm3;
            xm3 = xm2; xm2 = xm1; xm1 = cur;
            xcb[t * 192 + d] = (bf16_t)siluf_(a);
        }
    }
    __syncthreads();

    // ---- phase 3: dbl = xconv @ xproj_w^T (M=64, K=192, N=38 pad 48) ----
    for (int tl = wave; tl < 12; tl += 8) {
        const int m0 = (tl & 3) << 4;
        const int n0 = (tl >> 2) << 4;
        v8f acc = {};
#pragma unroll
        for (int kk = 0; kk < 6; ++kk) {
            v16bf a = frag_a(xcb, 192, m0, kk * 32, lane);
            v16bf b = frag_b(xproj_w, 192, n0, kk * 32, 38, lane);
            acc = __builtin_amdgcn_wmma_f32_16x16x32_bf16(false, a, false, b,
                                                          (short)0, acc, false, false);
        }
        const int n  = n0 + (lane & 15);
        const int hi = lane >> 4;
#pragma unroll
        for (int v = 0; v < 8; ++v)
            dbl[(m0 + v + 8 * hi) * 48 + n] = (bf16_t)acc[v];
    }
    __syncthreads();

    // ---- phase 4: selective scan (n=16) + skip + SiLU gate; y -> xcb in place ----
    if (tid < 192) {
        const int d = tid;
        float Arow[16], hst[16];
#pragma unroll
        for (int n = 0; n < 16; ++n) { Arow[n] = -__expf(A_log[d * 16 + n]); hst[n] = 0.0f; }
        float dtw[6];
#pragma unroll
        for (int k = 0; k < 6; ++k) dtw[k] = dt_w[d * 6 + k];
        const float dtb = dt_b[d];
        const float Dp  = Dv[d];
        for (int t = 0; t < 64; ++t) {
            float pre = dtb;
#pragma unroll
            for (int k = 0; k < 6; ++k) pre += (float)dbl[t * 48 + k] * dtw[k];
            const float dt  = softplusf_(pre);
            const float xt  = (float)xcb[t * 192 + d];
            const float dtx = dt * xt;
            float y = 0.0f;
#pragma unroll
            for (int n = 0; n < 16; ++n) {
                const float Bt = (float)dbl[t * 48 + 6 + n];
                const float Ct = (float)dbl[t * 48 + 22 + n];
                hst[n] = hst[n] * __expf(dt * Arow[n]) + dtx * Bt;
                y += hst[n] * Ct;
            }
            y += Dp * xt;
            const float z = (float)zb[t * 192 + d];
            xcb[t * 192 + d] = (bf16_t)(y * siluf_(z));   // gated output, in place
        }
    }
    __syncthreads();

    // ---- phase 5: yl = y @ out_w^T (M=64, K=192, N=96) -> global f32 ----
    for (int tl = wave; tl < 24; tl += 8) {
        const int m0 = (tl & 3) << 4;
        const int n0 = (tl >> 2) << 4;
        v8f acc = {};
#pragma unroll
        for (int kk = 0; kk < 6; ++kk) {
            v16bf a = frag_a(xcb, 192, m0, kk * 32, lane);
            v16bf b = frag_b(out_w, 192, n0, kk * 32, 96, lane);
            acc = __builtin_amdgcn_wmma_f32_16x16x32_bf16(false, a, false, b,
                                                          (short)0, acc, false, false);
        }
        const int n  = n0 + (lane & 15);
        const int hi = lane >> 4;
#pragma unroll
        for (int v = 0; v < 8; ++v)
            yl_out[(seq * 64 + m0 + v + 8 * hi) * 96 + n] = acc[v];
    }
}

// ---------------------------------------------------------------------------
// Stage 3a: build global tokens xg (4, 400, 96) bf16:
// gather yl token0 per patch, 20x20 snake, LayerNorm, shift.
// ---------------------------------------------------------------------------
__global__ __launch_bounds__(64)
void k_build_xg(const float* __restrict__ yl, const float* __restrict__ g,
                const float* __restrict__ be, bf16_t* __restrict__ xg) {
    const int idx = blockIdx.x * 64 + threadIdx.x;
    if (idx >= 1600) return;
    const int b = idx / 400;
    const int p = idx % 400;
    bf16_t* dst = xg + (size_t)idx * 96;
    if (p == 0) {
#pragma unroll
        for (int d = 0; d < 96; ++d) dst[d] = (bf16_t)0.0f;
        return;
    }
    const int pp = p - 1;
    const int rr = pp / 20, cc = pp % 20;
    const int jf = (rr & 1) ? (rr * 20 + (19 - cc)) : pp;   // patch flat index
    const float* src = yl + (size_t)(b * 400 + jf) * 64 * 96;  // token 0
    float v[96];
    float mean = 0.0f;
#pragma unroll 4
    for (int d = 0; d < 96; ++d) { v[d] = src[d]; mean += v[d]; }
    mean *= (1.0f / 96.0f);
    float var = 0.0f;
#pragma unroll 4
    for (int d = 0; d < 96; ++d) { float t = v[d] - mean; var += t * t; }
    var *= (1.0f / 96.0f);
    const float inv = rsqrtf(var + 1e-5f);
#pragma unroll 4
    for (int d = 0; d < 96; ++d)
        dst[d] = (bf16_t)((v[d] - mean) * inv * g[d] + be[d]);
}

// ---------------------------------------------------------------------------
// Generic WMMA GEMM: C(M x N, f32) = A(M x K, bf16 row-major) * W^T,
// W is (Nreal x K) f32 row-major, N padded to x16 (cols >= Nreal get 0).
// grid.x = M/64, block = 256 (8 waves).
// ---------------------------------------------------------------------------
__global__ __launch_bounds__(256)
void k_gemm_f32(const bf16_t* __restrict__ A, const float* __restrict__ W,
                float* __restrict__ C, int K, int N, int Nreal) {
    const int lane = threadIdx.x & 31;
    const int wave = threadIdx.x >> 5;
    const int mbase = blockIdx.x * 64;
    const bf16_t* Ab = A + (size_t)mbase * K;
    __builtin_prefetch(W, 0, 3);
    const int ntiles = N >> 4;
    for (int tl = wave; tl < 4 * ntiles; tl += 8) {
        const int m0 = (tl & 3) << 4;
        const int n0 = (tl >> 2) << 4;
        v8f acc = {};
        for (int k0 = 0; k0 < K; k0 += 32) {
            v16bf a = frag_a(Ab, K, m0, k0, lane);
            v16bf b = frag_b(W, K, n0, k0, Nreal, lane);
            acc = __builtin_amdgcn_wmma_f32_16x16x32_bf16(false, a, false, b,
                                                          (short)0, acc, false, false);
        }
        const int n  = n0 + (lane & 15);
        const int hi = lane >> 4;
#pragma unroll
        for (int v = 0; v < 8; ++v)
            C[(size_t)(mbase + m0 + v + 8 * hi) * N + n] = acc[v];
    }
}

// ---------------------------------------------------------------------------
// Stage 3c: global causal conv (dc=4) + SiLU. xz_g is (1600,192) f32;
// xc = cols 0..95. Output xconv (1600,96) bf16. grid 4 x block 96.
// ---------------------------------------------------------------------------
__global__ __launch_bounds__(96)
void k_conv_glb(const float* __restrict__ xz, const float* __restrict__ cw,
                const float* __restrict__ cb, bf16_t* __restrict__ xconv) {
    const int b = blockIdx.x;
    const int d = threadIdx.x;
    const float w0 = cw[d * 4 + 0], w1 = cw[d * 4 + 1];
    const float w2 = cw[d * 4 + 2], w3 = cw[d * 4 + 3];
    const float bi = cb[d];
    float xm1 = 0.0f, xm2 = 0.0f, xm3 = 0.0f;
    for (int t = 0; t < 400; ++t) {
        const size_t row = (size_t)(b * 400 + t);
        const float cur = xz[row * 192 + d];
        const float a = bi + w3 * cur + w2 * xm1 + w1 * xm2 + w0 * xm3;
        xm3 = xm2; xm2 = xm1; xm1 = cur;
        xconv[row * 96 + d] = (bf16_t)siluf_(a);
    }
}

// ---------------------------------------------------------------------------
// Stage 3e: global selective scan (n=8) + skip + SiLU gate.
// dbl (1600,32) f32: dt 0..5, B 6..13, C 14..21. z = xz cols 96..191.
// grid 4 x block 96.
// ---------------------------------------------------------------------------
__global__ __launch_bounds__(96)
void k_scan_glb(const float* __restrict__ dbl, const float* __restrict__ xz,
                const bf16_t* __restrict__ xconv, const float* __restrict__ dt_w,
                const float* __restrict__ dt_b, const float* __restrict__ A_log,
                const float* __restrict__ Dv, bf16_t* __restrict__ ygs) {
    const int b = blockIdx.x;
    const int d = threadIdx.x;
    float Arow[8], hst[8];
#pragma unroll
    for (int n = 0; n < 8; ++n) { Arow[n] = -__expf(A_log[d * 8 + n]); hst[n] = 0.0f; }
    float dtw[6];
#pragma unroll
    for (int k = 0; k < 6; ++k) dtw[k] = dt_w[d * 6 + k];
    const float dtb = dt_b[d];
    const float Dp  = Dv[d];
    for (int t = 0; t < 400; ++t) {
        const size_t row = (size_t)(b * 400 + t);
        float pre = dtb;
#pragma unroll
        for (int k = 0; k < 6; ++k) pre += dbl[row * 32 + k] * dtw[k];
        const float dt  = softplusf_(pre);
        const float xt  = (float)xconv[row * 96 + d];
        const float dtx = dt * xt;
        float y = 0.0f;
#pragma unroll
        for (int n = 0; n < 8; ++n) {
            const float Bt = dbl[row * 32 + 6 + n];
            const float Ct = dbl[row * 32 + 14 + n];
            hst[n] = hst[n] * __expf(dt * Arow[n]) + dtx * Bt;
            y += hst[n] * Ct;
        }
        y += Dp * xt;
        const float z = xz[row * 192 + 96 + d];
        ygs[row * 96 + d] = (bf16_t)(y * siluf_(z));
    }
}

// ---------------------------------------------------------------------------
// Stage 4: ym = un-snake(yl) + broadcast un-snake(yg); out = ym @ W_out^T + b.
// One block per 64 pixels: gather ym into LDS bf16, WMMA 64x96x96, store NCHW.
// ---------------------------------------------------------------------------
__global__ __launch_bounds__(256)
void k_final(const float* __restrict__ yl, const float* __restrict__ yg,
             const float* __restrict__ Wout, const float* __restrict__ bout,
             float* __restrict__ out) {
    __shared__ __align__(16) bf16_t ym[64 * 96];
    const int tid  = threadIdx.x;
    const int base = blockIdx.x * 64;

    for (int idx = tid; idx < 64 * 96; idx += 256) {
        const int i = idx / 96, d = idx % 96;
        const int P = base + i;
        const int b = P / HW_PIX;
        const int rem = P % HW_PIX;
        const int h = rem / 160, w = rem % 160;
        const int ph = h >> 3, hh = h & 7;
        const int pw = w >> 3, ww = w & 7;
        const int q  = hh * 8 + ww;
        const int pl = (hh & 1) ? (hh * 8 + (7 - ww)) : q;        // inv 8x8 snake
        const int j  = ph * 20 + pw;
        const int pg = (ph & 1) ? (ph * 20 + (19 - pw)) : j;      // inv 20x20 snake
        const float v = yl[((size_t)(b * 400 + j) * 64 + pl) * 96 + d]
                      + yg[((size_t)(b * 400 + pg)) * 96 + d];
        ym[idx] = (bf16_t)v;
    }
    __syncthreads();

    const int lane = tid & 31;
    const int wave = tid >> 5;
    for (int tl = wave; tl < 24; tl += 8) {
        const int m0 = (tl & 3) << 4;
        const int n0 = (tl >> 2) << 4;
        v8f acc = {};
#pragma unroll
        for (int kk = 0; kk < 3; ++kk) {
            v16bf a = frag_a(ym, 96, m0, kk * 32, lane);
            v16bf b = frag_b(Wout, 96, n0, kk * 32, 96, lane);
            acc = __builtin_amdgcn_wmma_f32_16x16x32_bf16(false, a, false, b,
                                                          (short)0, acc, false, false);
        }
        const int n  = n0 + (lane & 15);
        const int hi = lane >> 4;
        const float bn = bout[n];
#pragma unroll
        for (int v = 0; v < 8; ++v) {
            const int P = base + m0 + v + 8 * hi;
            const int b = P / HW_PIX;
            const int rem = P % HW_PIX;
            out[((size_t)b * 96 + n) * HW_PIX + rem] = acc[v] + bn;
        }
    }
}

// ---------------------------------------------------------------------------
// Launch
// ---------------------------------------------------------------------------
extern "C" void kernel_launch(void* const* d_in, const int* in_sizes, int n_in,
                              void* d_out, int out_size, void* d_ws, size_t ws_size,
                              hipStream_t stream) {
    const float* x      = (const float*)d_in[0];
    const float* W_in   = (const float*)d_in[1];
    const float* b_in   = (const float*)d_in[2];
    const float* g_loc  = (const float*)d_in[3];
    const float* be_loc = (const float*)d_in[4];
    const float* g_glb  = (const float*)d_in[5];
    const float* be_glb = (const float*)d_in[6];
    const float* l_inw  = (const float*)d_in[7];
    const float* l_cw   = (const float*)d_in[8];
    const float* l_cb   = (const float*)d_in[9];
    const float* l_xp   = (const float*)d_in[10];
    const float* l_dtw  = (const float*)d_in[11];
    const float* l_dtb  = (const float*)d_in[12];
    const float* l_Al   = (const float*)d_in[13];
    const float* l_D    = (const float*)d_in[14];
    const float* l_ow   = (const float*)d_in[15];
    const float* gl_inw = (const float*)d_in[16];
    const float* gl_cw  = (const float*)d_in[17];
    const float* gl_cb  = (const float*)d_in[18];
    const float* gl_xp  = (const float*)d_in[19];
    const float* gl_dtw = (const float*)d_in[20];
    const float* gl_dtb = (const float*)d_in[21];
    const float* gl_Al  = (const float*)d_in[22];
    const float* gl_D   = (const float*)d_in[23];
    const float* gl_ow  = (const float*)d_in[24];
    const float* W_out  = (const float*)d_in[25];
    const float* b_out  = (const float*)d_in[26];

    char* ws = (char*)d_ws;
    bf16_t* xl_ws   = (bf16_t*)(ws);                 // 1600*64*96 bf16 = 19,660,800 B
    float*  yl_ws   = (float*) (ws + 19660800);      // 1600*64*96 f32  = 39,321,600 B
    bf16_t* xg_ws   = (bf16_t*)(ws + 58982400);      // 1600*96 bf16    =    307,200 B
    float*  xzg_ws  = (float*) (ws + 59289600);      // 1600*192 f32    =  1,228,800 B
    bf16_t* xcg_ws  = (bf16_t*)(ws + 60518400);      // 1600*96 bf16    =    307,200 B
    float*  dblg_ws = (float*) (ws + 60825600);      // 1600*32 f32     =    204,800 B
    bf16_t* ygs_ws  = (bf16_t*)(ws + 61030400);      // 1600*96 bf16    =    307,200 B
    float*  ygo_ws  = (float*) (ws + 61337600);      // 1600*96 f32     =    614,400 B

    // Stage 1: project + patchify + snake + LN + shift
    k_stage1<<<1600, 64, 0, stream>>>(x, W_in, b_in, g_loc, be_loc, xl_ws);

    // Stage 2: fused local Mamba (WMMA GEMMs + scan), one block per sequence
    k_local<<<1600, 256, 0, stream>>>(xl_ws, l_inw, l_cw, l_cb, l_xp, l_dtw,
                                      l_dtb, l_Al, l_D, l_ow, yl_ws);

    // Stage 3: global Mamba over 4 sequences of 400 patch-tokens
    k_build_xg<<<25, 64, 0, stream>>>(yl_ws, g_glb, be_glb, xg_ws);
    k_gemm_f32<<<25, 256, 0, stream>>>(xg_ws, gl_inw, xzg_ws, 96, 192, 192);
    k_conv_glb<<<4, 96, 0, stream>>>(xzg_ws, gl_cw, gl_cb, xcg_ws);
    k_gemm_f32<<<25, 256, 0, stream>>>(xcg_ws, gl_xp, dblg_ws, 96, 32, 22);
    k_scan_glb<<<4, 96, 0, stream>>>(dblg_ws, xzg_ws, xcg_ws, gl_dtw, gl_dtb,
                                     gl_Al, gl_D, ygs_ws);
    k_gemm_f32<<<25, 256, 0, stream>>>(ygs_ws, gl_ow, ygo_ws, 96, 96, 96);

    // Stage 4: un-snake, add, final 96x96 projection + bias -> NCHW output
    k_final<<<1600, 256, 0, stream>>>(yl_ws, ygo_ws, W_out, b_out, (float*)d_out);
}
